// RecurrentProcessorLayerNorm_55972013802004
// MI455X (gfx1250) — matrix operand order repaired and checked by reading, not compiled, not run on hardware
//
#include <hip/hip_runtime.h>
#include <hip/hip_bf16.h>

typedef __attribute__((ext_vector_type(16))) __bf16 v16bf;
typedef __attribute__((ext_vector_type(8)))  float  v8f;
typedef unsigned int u32;

constexpr int B = 128, T = 64, L = 2, V = 32000, E = 768, H = 768;
constexpr int G = 4 * H;                 // 3072
constexpr float EPS = 1e-5f;

// LDS tile geometry: rows of 32 bf16 (64B) padded to 80B for 16B-aligned async writes.
constexpr int ROWB = 80;                 // bytes per LDS tile row
constexpr int ROWE = 40;                 // bf16 elements per LDS tile row (incl. pad)
constexpr int ROWU = 20;                 // u32 per LDS tile row

__device__ __forceinline__ float sigmoidf_(float x) { return 1.0f / (1.0f + __expf(-x)); }

// CDNA5 async DMA: global -> LDS, 16B per lane, tracked by ASYNCcnt.
__device__ __forceinline__ void async_b128(u32 lds_off, const void* g) {
  asm volatile("global_load_async_to_lds_b128 %0, %1, off"
               :: "v"(lds_off), "v"((unsigned long long)(size_t)g)
               : "memory");
}
__device__ __forceinline__ void wait_async0() {
  asm volatile("s_wait_asynccnt 0" ::: "memory");
}
__device__ __forceinline__ u32 lds_off_of(const void* shared_ptr) {
  // flat address of a __shared__ object: low 32 bits are the wave-relative LDS offset
  return (u32)(size_t)shared_ptr;
}

// ---------------------------------------------------------------- weight cast
__global__ __launch_bounds__(256) void k_cast(const float* __restrict__ wih,
                                              const float* __restrict__ whh,
                                              __bf16* __restrict__ wihb,
                                              __bf16* __restrict__ whhb) {
  const long long N1 = (long long)L * G * E;           // == L*G*H
  for (long long i = (long long)blockIdx.x * 256 + threadIdx.x; i < 2 * N1;
       i += (long long)gridDim.x * 256) {
    if (i < N1) wihb[i] = (__bf16)wih[i];
    else        whhb[i - N1] = (__bf16)whh[i - N1];
  }
}

// ---------------------------------------------------------------- embedding gather
__global__ __launch_bounds__(256) void k_embed(const long long* __restrict__ msg,
                                               const float* __restrict__ table,
                                               __bf16* __restrict__ emb) {
  const int r = blockIdx.x;                 // row = b*T + t
  long long tok = msg[r];
  if (tok < 0) tok = 0;
  if (tok >= V) tok = V - 1;
  const float* src = table + (size_t)tok * E;
  __bf16* dst = emb + (size_t)r * E;
#pragma unroll
  for (int j = 0; j < 3; ++j) {
    const int e = threadIdx.x + j * 256;
    dst[e] = (__bf16)src[e];
  }
}

// ---------------------------------------------------------------- state init
__global__ __launch_bounds__(256) void k_init(u32* __restrict__ hbf_u, float* __restrict__ cst) {
  const int NH = L * B * H / 2;   // bf16 pairs
  const int NC = L * B * H;
  for (int i = blockIdx.x * 256 + threadIdx.x; i < NC; i += gridDim.x * 256) {
    if (i < NH) hbf_u[i] = 0u;
    cst[i] = 0.0f;
  }
}

// ---------------------------------------------------------------- big GEMM: x_proj
// C[m, n] = sum_k emb[m,k] * Wih[n,k] + b_ih[n];  m=b*T+t in [0,8192), n=l*G+g in [0,6144)
// output layout: xproj[((t*L + l)*B + b)*G + g]
__global__ __launch_bounds__(256) void k_xproj(const __bf16* __restrict__ Aemb,
                                               const __bf16* __restrict__ Wih,
                                               const float* __restrict__ bih,
                                               float* __restrict__ xproj) {
  __shared__ u32 sA[2][128 * ROWU];
  __shared__ u32 sB[2][128 * ROWU];
  const int tid = threadIdx.x, lane = tid & 31, wv = tid >> 5;
  const int wm = wv >> 1, wn = wv & 1;            // 4x2 waves, each 32x64
  const int m0 = blockIdx.y * 128, n0 = blockIdx.x * 128;
  const char* Ag8 = (const char*)Aemb;
  const char* Bg8 = (const char*)Wih;
  const u32 ldsA = lds_off_of(&sA[0][0]);
  const u32 ldsB = lds_off_of(&sB[0][0]);

  v8f acc[2][4];
#pragma unroll
  for (int i = 0; i < 2; ++i)
#pragma unroll
    for (int j = 0; j < 4; ++j)
      acc[i][j] = v8f{0.f, 0.f, 0.f, 0.f, 0.f, 0.f, 0.f, 0.f};

  // issue one K-chunk (128x32 A tile + 128x32 B tile) into buffer `buf`
  auto issue = [&](int k0, int buf) {
    const u32 boff = (u32)buf * (128 * ROWB);
#pragma unroll
    for (int i = 0; i < 2; ++i) {
      const int idx = tid + i * 256;          // 512 16B chunks per tile
      const int r = idx >> 2, c = idx & 3;
      async_b128(ldsA + boff + r * ROWB + c * 16,
                 Ag8 + (size_t)(m0 + r) * (E * 2) + k0 * 2 + c * 16);
      async_b128(ldsB + boff + r * ROWB + c * 16,
                 Bg8 + (size_t)(n0 + r) * (E * 2) + k0 * 2 + c * 16);
    }
  };

  issue(0, 0);
  constexpr int KI = E / 32;                  // 24
  for (int kk = 0; kk < KI; ++kk) {
    const int cur = kk & 1;
    wait_async0();
    __syncthreads();                          // tile[cur] complete, prev reads retired
    if (kk + 1 < KI) issue((kk + 1) * 32, cur ^ 1);

    const __bf16* As = (const __bf16*)&sA[cur][0];
    const __bf16* Bs = (const __bf16*)&sB[cur][0];

    v16bf af[2], bfr[4];
    const int kg = (lane >> 4) << 3;
#pragma unroll
    for (int mf = 0; mf < 2; ++mf) {
      const int m = wm * 32 + mf * 16 + (lane & 15);
#pragma unroll
      for (int v = 0; v < 8; ++v) {
        const int k = ((v & 4) ? 16 : 0) + kg + ((v & 3) << 1);
        af[mf][2 * v]     = As[m * ROWE + k];
        af[mf][2 * v + 1] = As[m * ROWE + k + 1];
      }
    }
    const int kb = (lane >> 4) << 4;
#pragma unroll
    for (int nf = 0; nf < 4; ++nf) {
      const int n = wn * 64 + nf * 16 + (lane & 15);
#pragma unroll
      for (int v = 0; v < 8; ++v) {
        const int k = ((v & 4) ? 16 : 0) + ((lane >> 4) << 3) + ((v & 3) << 1);
        bfr[nf][2 * v]     = Bs[n * ROWE + k];
        bfr[nf][2 * v + 1] = Bs[n * ROWE + k + 1];
      }
    }
#pragma unroll
    for (int mf = 0; mf < 2; ++mf)
#pragma unroll
      for (int nf = 0; nf < 4; ++nf)
        acc[mf][nf] = __builtin_amdgcn_wmma_f32_16x16x32_bf16(
            false, af[mf], false, bfr[nf], (short)0, acc[mf][nf], false, false);
  }

#pragma unroll
  for (int mf = 0; mf < 2; ++mf)
#pragma unroll
    for (int nf = 0; nf < 4; ++nf) {
      const int n = n0 + wn * 64 + nf * 16 + (lane & 15);
      const int l = (n >= G) ? 1 : 0;
      const int g = n - l * G;
      const float bias = bih[n];
#pragma unroll
      for (int v = 0; v < 8; ++v) {
        const int m = m0 + wm * 32 + mf * 16 + ((lane >> 4) << 3) + v;
        const int b = m >> 6, t = m & 63;
        xproj[(((size_t)(t * L + l) * B + b) * G) + g] = acc[mf][nf][v] + bias;
      }
    }
}

// ---------------------------------------------------------------- per-step GEMM
// gates[l][b][g] = xproj[t][l][b][g] + b_hh[l][g] + sum_k h[l][b][k]*Whh[l][g][k]
__global__ __launch_bounds__(256) void k_step_gemm(const __bf16* __restrict__ hbf,
                                                   const __bf16* __restrict__ Whh,
                                                   const float* __restrict__ bhh,
                                                   const float* __restrict__ xproj,
                                                   float* __restrict__ gates, int t) {
  __shared__ u32 sA[2][128 * ROWU];
  __shared__ u32 sB[2][64 * ROWU];
  const int tid = threadIdx.x, lane = tid & 31, wv = tid >> 5;
  const int wm = wv >> 1, wn = wv & 1;            // 4x2 waves, each 32x32
  const int l = blockIdx.y;
  const int n0 = blockIdx.x * 64;
  const char* Ag8 = (const char*)(hbf + (size_t)l * B * H);
  const char* Bg8 = (const char*)(Whh + (size_t)l * G * H);
  const u32 ldsA = lds_off_of(&sA[0][0]);
  const u32 ldsB = lds_off_of(&sB[0][0]);

  v8f acc[2][2];
#pragma unroll
  for (int mf = 0; mf < 2; ++mf)
#pragma unroll
    for (int nf = 0; nf < 2; ++nf) {
      const int g = n0 + wn * 32 + nf * 16 + (lane & 15);
      const float bias = bhh[l * G + g];
#pragma unroll
      for (int v = 0; v < 8; ++v) {
        const int m = wm * 32 + mf * 16 + ((lane >> 4) << 3) + v;
        acc[mf][nf][v] = xproj[((size_t)(t * L + l) * B + m) * G + g] + bias;
      }
    }

  auto issue = [&](int k0, int buf) {
#pragma unroll
    for (int i = 0; i < 2; ++i) {             // A: 512 chunks
      const int idx = tid + i * 256;
      const int r = idx >> 2, c = idx & 3;
      async_b128(ldsA + (u32)buf * (128 * ROWB) + r * ROWB + c * 16,
                 Ag8 + (size_t)r * (H * 2) + k0 * 2 + c * 16);
    }
    {                                         // B: 256 chunks
      const int r = tid >> 2, c = tid & 3;
      async_b128(ldsB + (u32)buf * (64 * ROWB) + r * ROWB + c * 16,
                 Bg8 + (size_t)(n0 + r) * (H * 2) + k0 * 2 + c * 16);
    }
  };

  issue(0, 0);
  constexpr int KI = H / 32;                  // 24
  for (int kk = 0; kk < KI; ++kk) {
    const int cur = kk & 1;
    wait_async0();
    __syncthreads();
    if (kk + 1 < KI) issue((kk + 1) * 32, cur ^ 1);

    const __bf16* As = (const __bf16*)&sA[cur][0];
    const __bf16* Bs = (const __bf16*)&sB[cur][0];

    v16bf af[2], bfr[2];
    const int kg = (lane >> 4) << 3;
#pragma unroll
    for (int mf = 0; mf < 2; ++mf) {
      const int m = wm * 32 + mf * 16 + (lane & 15);
#pragma unroll
      for (int v = 0; v < 8; ++v) {
        const int k = ((v & 4) ? 16 : 0) + kg + ((v & 3) << 1);
        af[mf][2 * v]     = As[m * ROWE + k];
        af[mf][2 * v + 1] = As[m * ROWE + k + 1];
      }
    }
    const int kb = (lane >> 4) << 4;
#pragma unroll
    for (int nf = 0; nf < 2; ++nf) {
      const int n = wn * 32 + nf * 16 + (lane & 15);
#pragma unroll
      for (int v = 0; v < 8; ++v) {
        const int k = ((v & 4) ? 16 : 0) + ((lane >> 4) << 3) + ((v & 3) << 1);
        bfr[nf][2 * v]     = Bs[n * ROWE + k];
        bfr[nf][2 * v + 1] = Bs[n * ROWE + k + 1];
      }
    }
#pragma unroll
    for (int mf = 0; mf < 2; ++mf)
#pragma unroll
      for (int nf = 0; nf < 2; ++nf)
        acc[mf][nf] = __builtin_amdgcn_wmma_f32_16x16x32_bf16(
            false, af[mf], false, bfr[nf], (short)0, acc[mf][nf], false, false);
  }

#pragma unroll
  for (int mf = 0; mf < 2; ++mf)
#pragma unroll
    for (int nf = 0; nf < 2; ++nf) {
      const int g = n0 + wn * 32 + nf * 16 + (lane & 15);
#pragma unroll
      for (int v = 0; v < 8; ++v) {
        const int m = wm * 32 + mf * 16 + ((lane >> 4) << 3) + v;
        gates[((size_t)(l * B + m)) * G + g] = acc[mf][nf][v];
      }
    }
}

// ---------------------------------------------------------------- LSTM cell + 2x layernorm
__device__ __forceinline__ float block_sum(float v, float* red, int tid) {
  red[tid] = v;
  __syncthreads();
#pragma unroll
  for (int s = 128; s > 0; s >>= 1) {
    if (tid < s) red[tid] += red[tid + s];
    __syncthreads();
  }
  const float r = red[0];
  __syncthreads();
  return r;
}

__global__ __launch_bounds__(256) void k_step_cell(const float* __restrict__ gates,
                                                   float* __restrict__ cst,
                                                   __bf16* __restrict__ hbf,
                                                   float* __restrict__ hidden_all,
                                                   const float* __restrict__ gh,
                                                   const float* __restrict__ bh,
                                                   const float* __restrict__ gc,
                                                   const float* __restrict__ bc, int t) {
  __shared__ float red[256];
  const int tid = threadIdx.x;
  const int l = blockIdx.x >> 7;   // / 128
  const int b = blockIdx.x & 127;
  const float* grow = gates + (size_t)(l * B + b) * G;
  float* crow = cst + (size_t)(l * B + b) * H;

  float hv[3], cv[3];
#pragma unroll
  for (int j = 0; j < 3; ++j) {
    const int hh = tid + j * 256;
    const float ig = grow[hh];
    const float fg = grow[H + hh];
    const float gg = grow[2 * H + hh];
    const float og = grow[3 * H + hh];
    const float ct = sigmoidf_(fg) * crow[hh] + sigmoidf_(ig) * tanhf(gg);
    const float ht = sigmoidf_(og) * tanhf(ct);
    hv[j] = ht;
    cv[j] = ct;
  }
  const float inv = 1.0f / (float)H;
  const float mu_h = block_sum(hv[0] + hv[1] + hv[2], red, tid) * inv;
  float d0 = hv[0] - mu_h, d1 = hv[1] - mu_h, d2 = hv[2] - mu_h;
  const float var_h = block_sum(d0 * d0 + d1 * d1 + d2 * d2, red, tid) * inv;
  const float rs_h = rsqrtf(var_h + EPS);

  const float mu_c = block_sum(cv[0] + cv[1] + cv[2], red, tid) * inv;
  float e0 = cv[0] - mu_c, e1 = cv[1] - mu_c, e2 = cv[2] - mu_c;
  const float var_c = block_sum(e0 * e0 + e1 * e1 + e2 * e2, red, tid) * inv;
  const float rs_c = rsqrtf(var_c + EPS);

#pragma unroll
  for (int j = 0; j < 3; ++j) {
    const int hh = tid + j * 256;
    const float hl = (hv[j] - mu_h) * rs_h * gh[hh] + bh[hh];
    const float cl = (cv[j] - mu_c) * rs_c * gc[hh] + bc[hh];
    crow[hh] = cl;
    hbf[(size_t)(l * B + b) * H + hh] = (__bf16)hl;
    if (l == L - 1) hidden_all[((size_t)b * T + t) * H + hh] = hl;
  }
}

// ---------------------------------------------------------------- final gather
__global__ __launch_bounds__(256) void k_finalize(const float* __restrict__ hidden_all,
                                                  const long long* __restrict__ lens,
                                                  float* __restrict__ out) {
  const int b = blockIdx.x;
  long long idx = lens[b] - 1;
  if (idx < 0) idx = 0;
  if (idx >= T) idx = T - 1;
#pragma unroll
  for (int j = 0; j < 3; ++j) {
    const int hh = threadIdx.x + j * 256;
    out[(size_t)b * H + hh] = hidden_all[((size_t)b * T + (int)idx) * H + hh];
  }
}

// ---------------------------------------------------------------- launcher
extern "C" void kernel_launch(void* const* d_in, const int* in_sizes, int n_in,
                              void* d_out, int out_size, void* d_ws, size_t ws_size,
                              hipStream_t stream) {
  (void)in_sizes; (void)n_in; (void)out_size; (void)ws_size;
  const long long* msg   = (const long long*)d_in[0];
  const long long* lens  = (const long long*)d_in[1];
  const float*     table = (const float*)d_in[2];
  const float*     Wih   = (const float*)d_in[3];
  const float*     Whh   = (const float*)d_in[4];
  const float*     bih   = (const float*)d_in[5];
  const float*     bhh   = (const float*)d_in[6];
  const float*     gh    = (const float*)d_in[7];
  const float*     bh    = (const float*)d_in[8];
  const float*     gc    = (const float*)d_in[9];
  const float*     bc    = (const float*)d_in[10];
  float* out = (float*)d_out;

  char* ws = (char*)d_ws;
  size_t off = 0;
  auto carve = [&](size_t bytes) -> void* {
    void* p = ws + off;
    off = (off + bytes + 255) & ~(size_t)255;
    return p;
  };
  __bf16* emb   = (__bf16*)carve((size_t)B * T * E * 2);
  __bf16* wihb  = (__bf16*)carve((size_t)L * G * E * 2);
  __bf16* whhb  = (__bf16*)carve((size_t)L * G * H * 2);
  float*  xproj = (float*)carve((size_t)T * L * B * G * 4);
  __bf16* hbf   = (__bf16*)carve((size_t)L * B * H * 2);
  float*  cst   = (float*)carve((size_t)L * B * H * 4);
  float*  gates = (float*)carve((size_t)L * B * G * 4);
  float*  hid   = (float*)carve((size_t)B * T * H * 4);

  k_cast<<<2048, 256, 0, stream>>>(Wih, Whh, wihb, whhb);
  k_embed<<<B * T, 256, 0, stream>>>(msg, table, emb);
  k_init<<<768, 256, 0, stream>>>((u32*)hbf, cst);
  k_xproj<<<dim3(48, 64), 256, 0, stream>>>(emb, wihb, bih, xproj);

  for (int t = 0; t < T; ++t) {
    k_step_gemm<<<dim3(48, 2), 256, 0, stream>>>(hbf, whhb, bhh, xproj, gates, t);
    k_step_cell<<<L * B, 256, 0, stream>>>(gates, cst, hbf, hid, gh, bh, gc, bc, t);
  }
  k_finalize<<<B, 256, 0, stream>>>(hid, lens, out);
}
